// Attention_4466765988518
// MI455X (gfx1250) — compile-verified, matrix operand order checked by reading, not run on hardware
//
#include <hip/hip_runtime.h>
#include <hip/hip_bf16.h>
#include <stdint.h>

// ---------------------------------------------------------------------------
// Causal multi-head attention for MI455X (gfx1250, wave32, WMMA bf16)
// B=4, T=2048, C=1024, H=16, D=64
// Round 2: async Global->LDS staging (global_load_async_to_lds_b128 +
// s_wait_asynccnt) with double buffering; 32x64 wave tiles (8 WMMA / K-step).
// ---------------------------------------------------------------------------

typedef __bf16 v16bf __attribute__((ext_vector_type(16)));
typedef float  v8f   __attribute__((ext_vector_type(8)));

#define B_ 4
#define T_ 2048
#define C_ 1024
#define H_ 16
#define D_ 64
#define NEGINF (-1e30f)

union Frag16 { v16bf b; uint32_t u[8]; uint16_t h[16]; };

__device__ __forceinline__ uint16_t f2bf(float f) {
    union { float f; uint32_t u; } cv; cv.f = f;
    uint32_t u = cv.u;
    u += 0x7FFFu + ((u >> 16) & 1u);   // round-to-nearest-even
    return (uint16_t)(u >> 16);
}

// K index of the even element of packed pair p (0..7) in a 16-bit A/B fragment
// (ISA 7.12.2: VGPR v<4 -> K = half*8 + 2v + j ; v>=4 -> K = 16 + half*8 + 2(v-4) + j)
__device__ __forceinline__ int kPair(int p, int half) {
    return ((p >> 2) << 4) + (half << 3) + ((p & 3) << 1);
}

// --- CDNA5 async Global->LDS copy helpers -----------------------------------
// the LDS destination comes from the vdst VGPR (plus offset); low 32 bits of a generic pointer
// to __shared__ are the wave-relative LDS byte offset (ISA 10.2 aperture map).
__device__ __forceinline__ uint32_t lds_off(const void* p) {
    return (uint32_t)(uintptr_t)p;
}
__device__ __forceinline__ void async_cp_b128(uint32_t lds_byte_off, const void* gptr) {
    asm volatile("global_load_async_to_lds_b128 %0, %1, off"
                 :: "v"(lds_byte_off), "v"((uint64_t)(uintptr_t)gptr)
                 : "memory");
}
__device__ __forceinline__ void wait_async0() {
    asm volatile("s_wait_asynccnt 0" ::: "memory");
}

// ---------------------------------------------------------------------------
// fp32 -> bf16 conversion
// ---------------------------------------------------------------------------
__global__ __launch_bounds__(256) void cvt_bf16_kernel(
    const float* __restrict__ in, uint16_t* __restrict__ out, int n)
{
    int i = blockIdx.x * blockDim.x + threadIdx.x;
    int stride = gridDim.x * blockDim.x;
    for (; i < n; i += stride) out[i] = f2bf(in[i]);
}

// ---------------------------------------------------------------------------
// GEMM mainloop: block tile 128x128, 8 waves, wave tile 32(M) x 64(N).
// acc[msub*4+sub] covers rows wr*32+msub*16.., cols wc*64+sub*16..
// A[M,K] @ B[N,K]^T, both bf16 row-major with leading dim Kdim.
// Double-buffered async Global->LDS staging, 32-wide K steps.
// ---------------------------------------------------------------------------
#define TILE_DW 512   // 128 rows x 4 b128-chunks per tile

__device__ __forceinline__ void stage_tile_async(
    const uint8_t* __restrict__ A, const uint8_t* __restrict__ Bw,
    size_t rowBytes, int mBase, int nBase, int kOff,
    uint32_t asOff, uint32_t bsOff)
{
    const int tid = threadIdx.x;
    for (int i = 0; i < 2; ++i) {                 // A tile: 128x32 bf16 = 8KB
        int chunk = tid + (i << 8);               // 0..511
        int row = chunk >> 2, ch = chunk & 3;
        async_cp_b128(asOff + (chunk << 4),
                      A + (size_t)(mBase + row) * rowBytes + (size_t)kOff * 2 + (ch << 4));
    }
    for (int i = 0; i < 2; ++i) {                 // B tile: 128x32 bf16 = 8KB
        int chunk = tid + (i << 8);
        int row = chunk >> 2, ch = chunk & 3;
        async_cp_b128(bsOff + (chunk << 4),
                      Bw + (size_t)(nBase + row) * rowBytes + (size_t)kOff * 2 + (ch << 4));
    }
}

__device__ __forceinline__ void gemm_mainloop(
    const uint16_t* __restrict__ A, const uint16_t* __restrict__ Bw,
    int Kdim, int mBase, int nBase,
    uint16_t (*As)[128 * 32], uint16_t (*Bs)[128 * 32], v8f acc[8])
{
    const int lane = threadIdx.x & 31;
    const int wid  = threadIdx.x >> 5;
    const int wr   = wid & 3;      // M quarter (32 rows)
    const int wc   = wid >> 2;     // N half (64 cols)
    const int m    = lane & 15;
    const int half = lane >> 4;

    const v8f vzero = {0.f,0.f,0.f,0.f,0.f,0.f,0.f,0.f};
    for (int i = 0; i < 8; ++i) acc[i] = vzero;

    const uint8_t* A8 = (const uint8_t*)A;
    const uint8_t* B8 = (const uint8_t*)Bw;
    const size_t rowBytes = (size_t)Kdim * 2;
    const uint32_t asOff[2] = { lds_off(As[0]), lds_off(As[1]) };
    const uint32_t bsOff[2] = { lds_off(Bs[0]), lds_off(Bs[1]) };

    const int ksteps = Kdim >> 5;
    stage_tile_async(A8, B8, rowBytes, mBase, nBase, 0, asOff[0], bsOff[0]);

    for (int kk = 0; kk < ksteps; ++kk) {
        const int cur = kk & 1;
        wait_async0();          // this wave's slice of tile `cur` is in LDS
        __syncthreads();        // everyone's slice is in LDS; prev reads done
        if (kk + 1 < ksteps)
            stage_tile_async(A8, B8, rowBytes, mBase, nBase, (kk + 1) << 5,
                             asOff[cur ^ 1], bsOff[cur ^ 1]);

        const uint32_t* Asu = (const uint32_t*)As[cur];
        const uint32_t* Bsu = (const uint32_t*)Bs[cur];

        Frag16 bfr[4];
        for (int sub = 0; sub < 4; ++sub) {
            const uint32_t* br = Bsu + (wc * 64 + sub * 16 + m) * 16;
            for (int p = 0; p < 8; ++p) bfr[sub].u[p] = br[kPair(p, half) >> 1];
        }
        for (int msub = 0; msub < 2; ++msub) {
            Frag16 a;
            const uint32_t* ar = Asu + (wr * 32 + msub * 16 + m) * 16;
            for (int p = 0; p < 8; ++p) a.u[p] = ar[kPair(p, half) >> 1];
            for (int sub = 0; sub < 4; ++sub)
                acc[msub * 4 + sub] = __builtin_amdgcn_wmma_f32_16x16x32_bf16(
                    false, a.b, false, bfr[sub].b, (short)0, acc[msub * 4 + sub],
                    false, false);
        }
        // DS reads are consumed by WMMAs above (dscnt waited); the barrier at
        // the top of the next iteration fences tile reuse across waves.
    }
}

// ---------------------------------------------------------------------------
// Kernel: QKV projection, scatter to Q/K/V [b][h][t][d] bf16
// grid = (3072/128, 8192/128)
// ---------------------------------------------------------------------------
__global__ __launch_bounds__(256) void gemm_qkv_kernel(
    const uint16_t* __restrict__ xb, const uint16_t* __restrict__ wb,
    const float* __restrict__ bias, uint16_t* __restrict__ qkv)
{
    __shared__ __align__(16) uint16_t As[2][128 * 32];
    __shared__ __align__(16) uint16_t Bs[2][128 * 32];
    v8f acc[8];
    const int mBase = blockIdx.y * 128;
    const int nBase = blockIdx.x * 128;
    gemm_mainloop(xb, wb, C_, mBase, nBase, As, Bs, acc);

    const int lane = threadIdx.x & 31;
    const int wid  = threadIdx.x >> 5;
    const int wr = wid & 3, wc = wid >> 2;
    const int n = lane & 15, half = lane >> 4;
    const size_t S3 = (size_t)B_ * H_ * T_ * D_;
    for (int msub = 0; msub < 2; ++msub)
        for (int sub = 0; sub < 4; ++sub) {
            int col = nBase + wc * 64 + sub * 16 + n;
            int s   = col >> 10;          // 0:Q 1:K 2:V
            int rem = col & 1023;
            int h   = rem >> 6;
            int d   = rem & 63;
            float bv = bias[col];
            for (int r = 0; r < 8; ++r) {
                int row = mBase + wr * 32 + msub * 16 + half * 8 + r;
                int b = row >> 11, t = row & 2047;
                qkv[(size_t)s * S3 + (((size_t)(b * H_ + h) * T_ + t) * D_ + d)]
                    = f2bf(acc[msub * 4 + sub][r] + bv);
            }
        }
}

// ---------------------------------------------------------------------------
// Kernel: output projection, fp32 out + bias. grid = (1024/128, 8192/128)
// ---------------------------------------------------------------------------
__global__ __launch_bounds__(256) void gemm_out_kernel(
    const uint16_t* __restrict__ ob, const uint16_t* __restrict__ wb,
    const float* __restrict__ bias, float* __restrict__ out)
{
    __shared__ __align__(16) uint16_t As[2][128 * 32];
    __shared__ __align__(16) uint16_t Bs[2][128 * 32];
    v8f acc[8];
    const int mBase = blockIdx.y * 128;
    const int nBase = blockIdx.x * 128;
    gemm_mainloop(ob, wb, C_, mBase, nBase, As, Bs, acc);

    const int lane = threadIdx.x & 31;
    const int wid  = threadIdx.x >> 5;
    const int wr = wid & 3, wc = wid >> 2;
    const int n = lane & 15, half = lane >> 4;
    for (int msub = 0; msub < 2; ++msub)
        for (int sub = 0; sub < 4; ++sub) {
            int col = nBase + wc * 64 + sub * 16 + n;
            float bv = bias[col];
            for (int r = 0; r < 8; ++r) {
                int row = mBase + wr * 32 + msub * 16 + half * 8 + r;
                out[(size_t)row * C_ + col] = acc[msub * 4 + sub][r] + bv;
            }
        }
}

// ---------------------------------------------------------------------------
// Flash attention: one wave per 16-query tile, 32 keys per iteration.
// grid = B*H*(T/16)/8 = 1024 blocks of 256 threads (8 independent waves).
// ---------------------------------------------------------------------------
#define VT_STRIDE 36   // padded bf16 stride for transposed V rows (bank spread)

__global__ __launch_bounds__(256) void attn_kernel(
    const uint16_t* __restrict__ qw, const uint16_t* __restrict__ kw,
    const uint16_t* __restrict__ vw, uint16_t* __restrict__ ow)
{
    __shared__ uint16_t Plds[8][16 * 32];          // per-wave P tile (bf16)
    __shared__ uint16_t Vtlds[8][64 * VT_STRIDE];  // per-wave V^T tile (bf16)

    const int lane = threadIdx.x & 31;
    const int wid  = threadIdx.x >> 5;
    const int task = blockIdx.x * 8 + wid;         // 0 .. 8191
    const int qt = task & 127;                     // T/16 = 128
    const int bh = task >> 7;                      // 0 .. 63
    const int qBase = qt * 16;

    const int m = lane & 15;
    const int half = lane >> 4;
    const float scale = 0.125f;                    // 1/sqrt(64)

    // resident Q fragments (16 x 64, split into two K=32 fragments)
    Frag16 qa[2];
    {
        const uint32_t* qrow = (const uint32_t*)(qw + ((size_t)bh * T_ + (qBase + m)) * D_);
        for (int dh = 0; dh < 2; ++dh)
            for (int p = 0; p < 8; ++p)
                qa[dh].u[p] = qrow[(dh * 32 + kPair(p, half)) >> 1];
    }

    float mi[8], li[8], fr[8], p0[8], p1[8];
    v8f oacc[4];
    const v8f vzero = {0.f,0.f,0.f,0.f,0.f,0.f,0.f,0.f};
    for (int r = 0; r < 8; ++r) { mi[r] = NEGINF; li[r] = 0.f; }
    for (int i = 0; i < 4; ++i) oacc[i] = vzero;

    uint16_t* P  = Plds[wid];
    uint16_t* Vt = Vtlds[wid];

    const int nkb = (qBase + 16 + 31) >> 5;        // causal: key tiles needed
    for (int kb = 0; kb < nkb; ++kb) {
        const int kBase = kb << 5;

        // --- S = Q @ K^T (16 x 32), fp32 accum ---
        v8f s[2];
        for (int sub = 0; sub < 2; ++sub) {
            const uint32_t* krow =
                (const uint32_t*)(kw + ((size_t)bh * T_ + (kBase + sub * 16 + m)) * D_);
            Frag16 b0, b1;
            for (int p = 0; p < 8; ++p) {
                b0.u[p] = krow[kPair(p, half) >> 1];
                b1.u[p] = krow[(32 + kPair(p, half)) >> 1];
            }
            v8f a0 = __builtin_amdgcn_wmma_f32_16x16x32_bf16(
                false, qa[0].b, false, b0.b, (short)0, vzero, false, false);
            s[sub] = __builtin_amdgcn_wmma_f32_16x16x32_bf16(
                false, qa[1].b, false, b1.b, (short)0, a0, false, false);
        }

        // --- causal mask + online softmax (rows live across 16-lane halves) ---
        const int kc0 = kBase + m;
        for (int r = 0; r < 8; ++r) {
            int qrow_i = qBase + half * 8 + r;
            float v0 = s[0][r] * scale;
            float v1 = s[1][r] * scale;
            if (kc0 > qrow_i)      v0 = NEGINF;
            if (kc0 + 16 > qrow_i) v1 = NEGINF;
            float mx = fmaxf(v0, v1);
            for (int off = 8; off >= 1; off >>= 1)
                mx = fmaxf(mx, __shfl_xor(mx, off, 16));
            float mnew = fmaxf(mi[r], mx);
            float f  = __expf(mi[r] - mnew);
            float e0 = __expf(v0 - mnew);
            float e1 = __expf(v1 - mnew);
            float rs = e0 + e1;
            for (int off = 8; off >= 1; off >>= 1)
                rs += __shfl_xor(rs, off, 16);
            li[r] = li[r] * f + rs;
            mi[r] = mnew;
            fr[r] = f;
            p0[r] = e0; p1[r] = e1;
        }
        for (int i = 0; i < 4; ++i)
            for (int r = 0; r < 8; ++r)
                oacc[i][r] *= fr[r];

        // --- P (C-layout) -> LDS -> A-layout fragment ---
        for (int r = 0; r < 8; ++r) {
            P[(half * 8 + r) * 32 + m]      = f2bf(p0[r]);
            P[(half * 8 + r) * 32 + 16 + m] = f2bf(p1[r]);
        }
        // --- stage V tile transposed: Vt[d][kk] (wave-cooperative) ---
        {
            const uint32_t* vrow =
                (const uint32_t*)(vw + ((size_t)bh * T_ + (kBase + lane)) * D_);
            for (int p = 0; p < 32; ++p) {
                uint32_t w = vrow[p];
                Vt[(2 * p)     * VT_STRIDE + lane] = (uint16_t)(w & 0xFFFFu);
                Vt[(2 * p + 1) * VT_STRIDE + lane] = (uint16_t)(w >> 16);
            }
        }
        asm volatile("" ::: "memory");   // keep LDS write->read ordered (same wave, DS in-order)

        Frag16 pa;
        {
            const uint32_t* prow = (const uint32_t*)(P + m * 32);
            for (int p = 0; p < 8; ++p) pa.u[p] = prow[kPair(p, half) >> 1];
        }
        for (int ds = 0; ds < 4; ++ds) {
            Frag16 bv;
            const uint32_t* vtr = (const uint32_t*)(Vt + (ds * 16 + m) * VT_STRIDE);
            for (int p = 0; p < 8; ++p) bv.u[p] = vtr[kPair(p, half) >> 1];
            oacc[ds] = __builtin_amdgcn_wmma_f32_16x16x32_bf16(
                false, pa.b, false, bv.b, (short)0, oacc[ds], false, false);
        }
        asm volatile("" ::: "memory");
    }

    // --- normalize, scatter O as bf16 into [b][t][h*D+d] for the out-proj ---
    const int b = bh >> 4;
    const int h = bh & 15;
    for (int ds = 0; ds < 4; ++ds) {
        int d = ds * 16 + m;
        for (int r = 0; r < 8; ++r) {
            int t = qBase + half * 8 + r;
            ow[((size_t)(b * T_ + t)) * C_ + h * D_ + d] = f2bf(oacc[ds][r] / li[r]);
        }
    }
}

// ---------------------------------------------------------------------------
// Host-side launch
// ---------------------------------------------------------------------------
extern "C" void kernel_launch(void* const* d_in, const int* in_sizes, int n_in,
                              void* d_out, int out_size, void* d_ws, size_t ws_size,
                              hipStream_t stream)
{
    (void)in_sizes; (void)n_in; (void)out_size; (void)ws_size;

    const float* x    = (const float*)d_in[0];
    /* d_in[1] = causal mask: handled analytically */
    const float* Wqkv = (const float*)d_in[2];
    const float* bqkv = (const float*)d_in[3];
    const float* Wout = (const float*)d_in[4];
    const float* bout = (const float*)d_in[5];

    const size_t nX    = (size_t)B_ * T_ * C_;       // 8388608
    const size_t nWqkv = (size_t)3 * C_ * C_;        // 3145728
    const size_t nWout = (size_t)C_ * C_;            // 1048576
    const size_t S3    = (size_t)B_ * H_ * T_ * D_;  // 8388608

    uint8_t* ws = (uint8_t*)d_ws;
    uint16_t* xb    = (uint16_t*)ws;  ws += nX * 2;
    uint16_t* wqkvb = (uint16_t*)ws;  ws += nWqkv * 2;
    uint16_t* woutb = (uint16_t*)ws;  ws += nWout * 2;
    uint16_t* qkvw  = (uint16_t*)ws;  ws += 3 * S3 * 2;
    uint16_t* obuf  = (uint16_t*)ws;  ws += nX * 2;      // ~88 MB total

    cvt_bf16_kernel<<<4096, 256, 0, stream>>>(x,    xb,    (int)nX);
    cvt_bf16_kernel<<<4096, 256, 0, stream>>>(Wqkv, wqkvb, (int)nWqkv);
    cvt_bf16_kernel<<<2048, 256, 0, stream>>>(Wout, woutb, (int)nWout);

    gemm_qkv_kernel<<<dim3(3 * C_ / 128, (B_ * T_) / 128), 256, 0, stream>>>(
        xb, wqkvb, bqkv, qkvw);

    attn_kernel<<<(B_ * H_ * (T_ / 16)) / 8, 256, 0, stream>>>(
        qkvw, qkvw + S3, qkvw + 2 * S3, obuf);

    gemm_out_kernel<<<dim3(C_ / 128, (B_ * T_) / 128), 256, 0, stream>>>(
        obuf, woutb, bout, (float*)d_out);
}
